// MoE_51221779972403
// MI455X (gfx1250) — matrix operand order repaired and checked by reading, not compiled, not run on hardware
//
#include <hip/hip_runtime.h>

typedef _Float16 v16h __attribute__((ext_vector_type(16)));
typedef float    v8f  __attribute__((ext_vector_type(8)));

#define Bq   16
#define Nq   8192
#define Dq   64
#define Eq   4
#define Hq   128
#define GHq  64
#define TOK  (Bq * Nq)          // 131072 tokens
#define WPB  8                  // waves per block (256 threads, wave32)
#define TPW  2                  // 16-token tiles per wave
#define TOKPB (WPB * TPW * 16)  // 256 tokens per block
#define NBLK  (TOK / TOKPB)     // 512 blocks

__device__ __forceinline__ float softplusf(float v) {
  return (v > 20.0f) ? v : log1pf(expf(v));
}

__global__ __launch_bounds__(256) void moe_fused(
    const float* __restrict__ x,   const float* __restrict__ coarse,
    const float* __restrict__ Wg1, const float* __restrict__ bg1,
    const float* __restrict__ Wg2, const float* __restrict__ bg2,
    const float* __restrict__ We,  const float* __restrict__ be,
    const float* __restrict__ Wm,  const float* __restrict__ bm,
    const float* __restrict__ Wv,  const float* __restrict__ bv,
    float* __restrict__ refined, float* __restrict__ lossp,
    float* __restrict__ delta)
{
  // B-operand layouts, pre-swizzled: blob = [32 lanes][16 halves] (1 KB each).
  // Lane l of blob (nt,kb) holds B[k = kb*32 + (l>>4)*16 + i][n = nt*16 + (l&15)].
  __shared__ __align__(32) _Float16 ldsWe[64 * 512];   // 64 KB: 32 ntiles x 2 kblocks
  __shared__ __align__(32) _Float16 ldsWg1[8 * 512];   //  8 KB:  4 ntiles x 2 kblocks

  const int tid = threadIdx.x;

  // ---- Stage weights to LDS as f16 in WMMA-B layout (cooperative, once) ----
  for (int idx = tid; idx < 64 * 512; idx += 256) {
    int bi = idx >> 9, j = idx & 511;
    int ln = j >> 4, i = j & 15;
    int nt = bi >> 1, kb = bi & 1;
    int col = nt * 16 + (ln & 15);           // 0..511 = e*128 + h
    int k   = kb * 32 + (ln >> 4) * 16 + i;  // 0..63
    int e = col >> 7, h = col & 127;
    ldsWe[idx] = (_Float16)We[(e * Dq + k) * Hq + h];
  }
  for (int idx = tid; idx < 8 * 512; idx += 256) {
    int bi = idx >> 9, j = idx & 511;
    int ln = j >> 4, i = j & 15;
    int nt = bi >> 1, kb = bi & 1;
    int g = nt * 16 + (ln & 15);             // 0..63
    int k = kb * 32 + (ln >> 4) * 16 + i;    // 0..63
    ldsWg1[idx] = (_Float16)Wg1[k * GHq + g];
  }
  __syncthreads();

  const int wave = tid >> 5;
  const int lane = tid & 31;
  const int l16  = lane & 15;
  const int hi   = lane >> 4;

  const float bg2v0 = bg2[0], bg2v1 = bg2[1], bg2v2 = bg2[2], bg2v3 = bg2[3];

  float loss_part = 0.0f;

  for (int t = 0; t < TPW; ++t) {
    const int tile = blockIdx.x * (WPB * TPW) + wave * TPW + t;
    const int tok0 = tile * 16;

    // ---- Load x tile as f16 A-operands (16x32 layout, K=0..31 and K=32..63) ----
    // Lane l: M = l&15, K = hi*16 .. +15 (contiguous) per operand.
    const float* xr = x + (size_t)(tok0 + l16) * Dq + hi * 16;
    v16h a0, a1;
#pragma unroll
    for (int i = 0; i < 8; ++i) {
      a0[i]     = (_Float16)xr[i];
      a0[8 + i] = (_Float16)xr[8 + i];
      a1[i]     = (_Float16)xr[32 + i];
      a1[8 + i] = (_Float16)xr[40 + i];
    }

    // =============== Gate: h = relu(x@Wg1 + bg1); logits = h@Wg2 + bg2 ========
    float lacc[8][4];
#pragma unroll
    for (int r = 0; r < 8; ++r)
#pragma unroll
      for (int e = 0; e < 4; ++e) lacc[r][e] = 0.0f;

#pragma unroll
    for (int nt = 0; nt < 4; ++nt) {
      v8f c = {0.f, 0.f, 0.f, 0.f, 0.f, 0.f, 0.f, 0.f};
      const v16h b0 = *(const v16h*)(ldsWg1 + (nt * 2 + 0) * 512 + lane * 16);
      const v16h b1 = *(const v16h*)(ldsWg1 + (nt * 2 + 1) * 512 + lane * 16);
      c = __builtin_amdgcn_wmma_f32_16x16x32_f16(false, a0, false, b0, (short)0, c, false, false);
      c = __builtin_amdgcn_wmma_f32_16x16x32_f16(false, a1, false, b1, (short)0, c, false, false);
      const int g = nt * 16 + l16;           // hidden index held by this lane
      const float bias = bg1[g];
      const float w0 = Wg2[g * 4 + 0], w1 = Wg2[g * 4 + 1];
      const float w2 = Wg2[g * 4 + 2], w3 = Wg2[g * 4 + 3];
#pragma unroll
      for (int r = 0; r < 8; ++r) {          // token rows r (lanes 0-15) / r+8 (lanes 16-31)
        float f = fmaxf(c[r] + bias, 0.0f);
        lacc[r][0] = fmaf(f, w0, lacc[r][0]);
        lacc[r][1] = fmaf(f, w1, lacc[r][1]);
        lacc[r][2] = fmaf(f, w2, lacc[r][2]);
        lacc[r][3] = fmaf(f, w3, lacc[r][3]);
      }
    }
    // reduce partial dots across the 16-lane half (h dimension)
#pragma unroll
    for (int r = 0; r < 8; ++r)
#pragma unroll
      for (int e = 0; e < 4; ++e) {
        float v = lacc[r][e];
        v += __shfl_xor(v, 1, 32);
        v += __shfl_xor(v, 2, 32);
        v += __shfl_xor(v, 4, 32);
        v += __shfl_xor(v, 8, 32);
        lacc[r][e] = v;
      }

    int amax[8];
#pragma unroll
    for (int r = 0; r < 8; ++r) {
      const float l0 = lacc[r][0] + bg2v0, l1 = lacc[r][1] + bg2v1;
      const float l2 = lacc[r][2] + bg2v2, l3 = lacc[r][3] + bg2v3;
      int am = 0; float mx = l0;
      if (l1 > mx) { mx = l1; am = 1; }
      if (l2 > mx) { mx = l2; am = 2; }
      if (l3 > mx) { mx = l3; am = 3; }
      amax[r] = am;
      if (l16 == 0) {  // one writer lane per 16-lane half does softmax/entropy
        const float e0 = expf(l0 - mx), e1 = expf(l1 - mx);
        const float e2 = expf(l2 - mx), e3 = expf(l3 - mx);
        const float inv = 1.0f / (e0 + e1 + e2 + e3);
        const float p0 = e0 * inv, p1 = e1 * inv, p2 = e2 * inv, p3 = e3 * inv;
        loss_part -= p0 * logf(p0 + 1e-8f) + p1 * logf(p1 + 1e-8f) +
                     p2 * logf(p2 + 1e-8f) + p3 * logf(p3 + 1e-8f);
      }
    }

    // =============== Experts: feat = relu(x@We_e + be_e); out = feat@[Wm|Wv] ==
    float sel[8][4];
#pragma unroll
    for (int r = 0; r < 8; ++r)
#pragma unroll
      for (int o = 0; o < 4; ++o) sel[r][o] = 0.0f;

    for (int e = 0; e < Eq; ++e) {
      float macc[8][4];
#pragma unroll
      for (int r = 0; r < 8; ++r)
#pragma unroll
        for (int o = 0; o < 4; ++o) macc[r][o] = 0.0f;

#pragma unroll
      for (int n8 = 0; n8 < 8; ++n8) {
        const int nt = e * 8 + n8;
        v8f c = {0.f, 0.f, 0.f, 0.f, 0.f, 0.f, 0.f, 0.f};
        const v16h b0 = *(const v16h*)(ldsWe + (nt * 2 + 0) * 512 + lane * 16);
        const v16h b1 = *(const v16h*)(ldsWe + (nt * 2 + 1) * 512 + lane * 16);
        c = __builtin_amdgcn_wmma_f32_16x16x32_f16(false, a0, false, b0, (short)0, c, false, false);
        c = __builtin_amdgcn_wmma_f32_16x16x32_f16(false, a1, false, b1, (short)0, c, false, false);
        const int h = n8 * 16 + l16;
        const float bias = be[e * Hq + h];
        const float wm0 = Wm[(e * Hq + h) * 2 + 0], wm1 = Wm[(e * Hq + h) * 2 + 1];
        const float wv0 = Wv[(e * Hq + h) * 2 + 0], wv1 = Wv[(e * Hq + h) * 2 + 1];
#pragma unroll
        for (int r = 0; r < 8; ++r) {
          float f = fmaxf(c[r] + bias, 0.0f);
          macc[r][0] = fmaf(f, wm0, macc[r][0]);
          macc[r][1] = fmaf(f, wm1, macc[r][1]);
          macc[r][2] = fmaf(f, wv0, macc[r][2]);
          macc[r][3] = fmaf(f, wv1, macc[r][3]);
        }
      }
#pragma unroll
      for (int r = 0; r < 8; ++r)
#pragma unroll
        for (int o = 0; o < 4; ++o) {
          float v = macc[r][o];
          v += __shfl_xor(v, 1, 32);
          v += __shfl_xor(v, 2, 32);
          v += __shfl_xor(v, 4, 32);
          v += __shfl_xor(v, 8, 32);
          macc[r][o] = v;
        }
      const float bm0 = bm[e * 2 + 0], bm1 = bm[e * 2 + 1];
      const float bv0 = bv[e * 2 + 0], bv1 = bv[e * 2 + 1];
#pragma unroll
      for (int r = 0; r < 8; ++r) {
        const bool pick = (amax[r] == e);   // top-1, normalized weight == 1
        sel[r][0] = pick ? macc[r][0] + bm0 : sel[r][0];
        sel[r][1] = pick ? macc[r][1] + bm1 : sel[r][1];
        sel[r][2] = pick ? macc[r][2] + bv0 : sel[r][2];
        sel[r][3] = pick ? macc[r][3] + bv1 : sel[r][3];
      }
    }

    // =============== Nonlinearity + outputs (writer lanes 0 and 16) ===========
    if (l16 == 0) {
      const int tb = tok0 + hi * 8;
#pragma unroll
      for (int r = 0; r < 8; ++r) {
        const int tk = tb + r;
        const float m0 = tanhf(sel[r][0]);
        const float m1 = tanhf(sel[r][1]);
        const float v0 = softplusf(sel[r][2]);
        const float v1 = softplusf(sel[r][3]);
        delta[(size_t)tk * 4 + 0] = m0;
        delta[(size_t)tk * 4 + 1] = m1;
        delta[(size_t)tk * 4 + 2] = v0;
        delta[(size_t)tk * 4 + 3] = v1;
        float c0 = coarse[(size_t)tk * 2 + 0] + m0 * 0.002f;
        float c1 = coarse[(size_t)tk * 2 + 1] + m1 * 0.002f;
        c0 = fminf(fmaxf(c0, 0.0f), 1.0f);
        c1 = fminf(fmaxf(c1, 0.0f), 1.0f);
        refined[(size_t)tk * 2 + 0] = c0;
        refined[(size_t)tk * 2 + 1] = c1;
      }
    }
  }

  // loss: combine the two writer lanes, one atomic per wave
  loss_part += __shfl_xor(loss_part, 16, 32);
  if (lane == 0) atomicAdd(lossp, loss_part * (1.0f / (float)TOK));
}

extern "C" void kernel_launch(void* const* d_in, const int* in_sizes, int n_in,
                              void* d_out, int out_size, void* d_ws, size_t ws_size,
                              hipStream_t stream) {
  (void)in_sizes; (void)n_in; (void)out_size; (void)d_ws; (void)ws_size;
  const float* x      = (const float*)d_in[0];
  const float* coarse = (const float*)d_in[1];
  const float* Wg1    = (const float*)d_in[2];
  const float* bg1    = (const float*)d_in[3];
  const float* Wg2    = (const float*)d_in[4];
  const float* bg2    = (const float*)d_in[5];
  const float* We     = (const float*)d_in[6];
  const float* be     = (const float*)d_in[7];
  const float* Wm     = (const float*)d_in[8];
  const float* bm     = (const float*)d_in[9];
  const float* Wv     = (const float*)d_in[10];
  const float* bv     = (const float*)d_in[11];

  float* out     = (float*)d_out;
  float* refined = out;                       // [16,8192,2] = 262144 floats
  float* lossp   = out + (size_t)TOK * 2;     // 1 float
  float* deltap  = out + (size_t)TOK * 2 + 1; // [16,8192,4] = 524288 floats

  hipMemsetAsync(lossp, 0, sizeof(float), stream);  // graph-capturable memset node
  hipLaunchKernelGGL(moe_fused, dim3(NBLK), dim3(256), 0, stream,
                     x, coarse, Wg1, bg1, Wg2, bg2, We, be, Wm, bm, Wv, bv,
                     refined, lossp, deltap);
}
